// CaMoE_Block_61546881352172
// MI455X (gfx1250) — matrix operand order repaired and checked by reading, not compiled, not run on hardware
//
#include <hip/hip_runtime.h>

#define DEVFN __device__ __forceinline__

typedef __attribute__((ext_vector_type(16))) __bf16 v16bf;
typedef __attribute__((ext_vector_type(8)))  float  v8f;

union FragBF { uint4 q[2]; v16bf v; };

DEVFN unsigned short f2bf(float x) {
  unsigned u = __float_as_uint(x);
  unsigned r = (u + 0x7FFFu + ((u >> 16) & 1u)) >> 16;   // RNE
  return (unsigned short)r;
}
DEVFN unsigned pack2bf(float lo, float hi) {
  return (unsigned)f2bf(lo) | ((unsigned)f2bf(hi) << 16);
}
DEVFN float bf2f(unsigned short h) { return __uint_as_float(((unsigned)h) << 16); }
DEVFN float sigmf(float x) { return 1.0f / (1.0f + __expf(-x)); }
DEVFN float geluf(float x) {
  float x3 = x * x * x;
  return 0.5f * x * (1.0f + tanhf(0.7978845608028654f * (x + 0.044715f * x3)));
}

enum { EPI_NONE = 0, EPI_SIGMOID = 1, EPI_VMIX = 2, EPI_BRIDGE = 3,
       EPI_RELU2 = 4, EPI_GELU = 5, EPI_SIGMUL = 6 };

struct GemmP {
  const unsigned short* A; int lda;        // bf16 activations, row-major
  const float* Bw;         int ldb;        // f32 weights, row-major [K,N]
  int M, N, K;
  const int* gather;                        // A-row gather (perm) or null
  const int* cntp;                          // device count (effective M) or null
  const int* offp;                          // device entry offset or null
  float* outF; unsigned short* outH; int ldo;
  const float* aux0; const float* aux1;     // v_first / g_v / bias
  const unsigned short* hsrc; int ldh;      // bridge residual source
};

// 128x128 block tile, 256 threads = 8 waves (4M x 2N), wave tile 32x64.
// KT=64 per stage (2 WMMA K-steps), LDS ping-pong double buffer with
// register prefetch: one s_barrier per 64-K slab. Gather indices and all
// global addresses are hoisted out of the K loop. B tile is transposed
// f32->bf16 with paired-k packing so all DS stores are 32-bit.
template<int EPI>
__global__ __launch_bounds__(256) void gemm_wmma_bf16(GemmP p)
{
  const int Meff  = p.cntp ? p.cntp[0] : p.M;
  const int obase = p.offp ? p.offp[0] : 0;
  if ((int)blockIdx.y * 128 >= Meff) return;

  __shared__ __align__(16) unsigned short As[2][128 * 64];  // [row][k] bf16
  __shared__ __align__(16) unsigned short Bs[2][128 * 64];  // [col][k] bf16 (transposed)

  const int tid  = threadIdx.x;
  const int lane = tid & 31;
  const int wv   = tid >> 5;
  const int wm   = wv >> 1;       // 0..3
  const int wn   = wv & 1;        // 0..1

  // ---- hoisted per-thread global addresses ----
  // A: 4 x uint4 per stage; idx -> row = idx>>3 (0..127), kg = idx&7
  const unsigned short* aptr[4];
#pragma unroll
  for (int i = 0; i < 4; i++) {
    int idx  = tid + i * 256;
    int rowt = idx >> 3, kg = idx & 7;
    int mg   = blockIdx.y * 128 + rowt;
    if (mg < Meff) {
      int ar  = p.gather ? p.gather[obase + mg] : (obase + mg);
      aptr[i] = p.A + (size_t)ar * p.lda + kg * 8;
    } else aptr[i] = nullptr;
  }
  // B: 4 row-pairs of float4 per stage; idx -> pr = idx>>5 (0..31 -> k=2*pr),
  // ng = idx&31 (n-group of 4)
  const float* bptr[4];
#pragma unroll
  for (int i = 0; i < 4; i++) {
    int idx = tid + i * 256;
    int pr = idx >> 5, ng = idx & 31;
    bptr[i] = p.Bw + (size_t)(2 * pr) * p.ldb + blockIdx.x * 128 + ng * 4;
  }

  uint4  areg[4];
  float4 breg[8];                 // [2*i] = row 2pr, [2*i+1] = row 2pr+1
  auto fetch = [&](int kt) {
#pragma unroll
    for (int i = 0; i < 4; i++)
      areg[i] = aptr[i] ? *(const uint4*)(aptr[i] + kt) : make_uint4(0u, 0u, 0u, 0u);
#pragma unroll
    for (int i = 0; i < 4; i++) {
      const float* bp = bptr[i] + (size_t)kt * p.ldb;
      breg[2 * i]     = *(const float4*)bp;
      breg[2 * i + 1] = *(const float4*)(bp + p.ldb);
    }
  };
  auto stage = [&](int buf) {
    uint4* As4 = (uint4*)As[buf];
#pragma unroll
    for (int i = 0; i < 4; i++) As4[tid + i * 256] = areg[i];
    unsigned* Bs32 = (unsigned*)Bs[buf];
#pragma unroll
    for (int i = 0; i < 4; i++) {
      int idx = tid + i * 256;
      int pr = idx >> 5, ng = idx & 31;
      const float4 r0 = breg[2 * i], r1 = breg[2 * i + 1];
      // Bs[n][k] k-pair (2pr,2pr+1) packed into one dword at n*32 + pr
      Bs32[(ng * 4 + 0) * 32 + pr] = pack2bf(r0.x, r1.x);
      Bs32[(ng * 4 + 1) * 32 + pr] = pack2bf(r0.y, r1.y);
      Bs32[(ng * 4 + 2) * 32 + pr] = pack2bf(r0.z, r1.z);
      Bs32[(ng * 4 + 3) * 32 + pr] = pack2bf(r0.w, r1.w);
    }
  };

  v8f acc[8] = {};
  const int aoffq = (lane < 16) ? 0 : 2;     // uint4 index within 32-k slice
  const int arow  = wm * 32 + (lane & 15);
  const int boffq = (lane < 16) ? 0 : 2;
  const int bcol  = wn * 64 + (lane & 15);

  auto compute = [&](int buf) {
    const uint4* As4 = (const uint4*)As[buf];
    const uint4* Bs4 = (const uint4*)Bs[buf];
#pragma unroll
    for (int ks = 0; ks < 2; ks++) {
      FragBF fa[2], fb[4];
#pragma unroll
      for (int mi = 0; mi < 2; mi++) {
        int base = (arow + mi * 16) * 8 + ks * 4;   // 64 ushorts = 8 uint4 / row
        fa[mi].q[0] = As4[base + aoffq];
        fa[mi].q[1] = As4[base + aoffq + 1];
      }
#pragma unroll
      for (int ni = 0; ni < 4; ni++) {
        int base = (bcol + ni * 16) * 8 + ks * 4;
        fb[ni].q[0] = Bs4[base + boffq];
        fb[ni].q[1] = Bs4[base + boffq + 1];
      }
#pragma unroll
      for (int mi = 0; mi < 2; mi++)
#pragma unroll
        for (int ni = 0; ni < 4; ni++)
          acc[mi * 4 + ni] = __builtin_amdgcn_wmma_f32_16x16x32_bf16(
              false, fa[mi].v, false, fb[ni].v, (short)0, acc[mi * 4 + ni],
              false, false);
    }
  };

  const int nk = p.K >> 6;       // K is a multiple of 64 everywhere here
  fetch(0);
  stage(0);
  __syncthreads();
  for (int s = 0; s < nk; s++) {
    int buf = s & 1;
    if (s + 1 < nk) {
      fetch((s + 1) << 6);
      if (s + 2 < nk)
        __builtin_prefetch(bptr[0] + (size_t)((s + 2) << 6) * p.ldb, 0, 1);
    }
    compute(buf);
    if (s + 1 < nk) stage(buf ^ 1);
    __syncthreads();
  }

  // ---- epilogue: C/D layout -> VGPR j: row = base + j + (lane>=16)*8,
  //      col = base + (lane&15) ----
  const int colL = lane & 15;
  const int rAdd = (lane & 16) ? 8 : 0;
#pragma unroll
  for (int mi = 0; mi < 2; mi++)
#pragma unroll
    for (int ni = 0; ni < 4; ni++)
#pragma unroll
      for (int j = 0; j < 8; j++) {
        int rowg = blockIdx.y * 128 + wm * 32 + mi * 16 + rAdd + j;
        if (rowg >= Meff) continue;
        int colg = blockIdx.x * 128 + wn * 64 + ni * 16 + colL;
        float a = acc[mi * 4 + ni][j];
        size_t orow = (size_t)(obase + rowg);
        if constexpr (EPI == EPI_NONE) {
          p.outF[orow * p.ldo + colg] = a;
        } else if constexpr (EPI == EPI_SIGMOID) {
          p.outF[orow * p.ldo + colg] = sigmf(a);
        } else if constexpr (EPI == EPI_VMIX) {
          float vf = p.aux0[orow * p.ldo + colg];
          float g  = sigmf(p.aux1[colg]);
          p.outF[orow * p.ldo + colg] = a + (vf - a) * g;
        } else if constexpr (EPI == EPI_BRIDGE) {
          float h = bf2f(p.hsrc[orow * p.ldh + colg]);
          p.outH[orow * p.ldo + colg] = f2bf(a + p.aux0[colg] + h);
        } else if constexpr (EPI == EPI_RELU2) {
          float r = a > 0.f ? a : 0.f;
          p.outH[orow * p.ldo + colg] = f2bf(r * r);
        } else if constexpr (EPI == EPI_GELU) {
          p.outH[orow * p.ldo + colg] = f2bf(geluf(a));
        } else if constexpr (EPI == EPI_SIGMUL) {
          p.outF[orow * p.ldo + colg] *= sigmf(a);
        }
      }
}

// LayerNorm over C=1024; optional residual add; writes bf16 normalized row
// (into hcat at stride ldoH) and optionally the un-normalized f32 row.
__global__ __launch_bounds__(256) void ln_k(
    const float* __restrict__ x, const float* __restrict__ add,
    const float* __restrict__ g, const float* __restrict__ b,
    unsigned short* __restrict__ outH, int ldoH, float* __restrict__ outF)
{
  __shared__ float red[256];
  const int tok = blockIdx.x, t = threadIdx.x;
  float4 xv = ((const float4*)(x + (size_t)tok * 1024))[t];
  if (add) {
    float4 av = ((const float4*)(add + (size_t)tok * 1024))[t];
    xv.x += av.x; xv.y += av.y; xv.z += av.z; xv.w += av.w;
  }
  red[t] = xv.x + xv.y + xv.z + xv.w;
  __syncthreads();
  for (int o = 128; o > 0; o >>= 1) { if (t < o) red[t] += red[t + o]; __syncthreads(); }
  float mean = red[0] * (1.0f / 1024.0f);
  __syncthreads();
  float dx = xv.x - mean, dy = xv.y - mean, dz = xv.z - mean, dw = xv.w - mean;
  red[t] = dx * dx + dy * dy + dz * dz + dw * dw;
  __syncthreads();
  for (int o = 128; o > 0; o >>= 1) { if (t < o) red[t] += red[t + o]; __syncthreads(); }
  float rstd = rsqrtf(red[0] * (1.0f / 1024.0f) + 1e-5f);
  if (outF) ((float4*)(outF + (size_t)tok * 1024))[t] = xv;
  float4 gv = ((const float4*)g)[t], bv = ((const float4*)b)[t];
  unsigned short* o = outH + (size_t)tok * ldoH + t * 4;
  o[0] = f2bf(dx * rstd * gv.x + bv.x);
  o[1] = f2bf(dy * rstd * gv.y + bv.y);
  o[2] = f2bf(dz * rstd * gv.z + bv.z);
  o[3] = f2bf(dw * rstd * gv.w + bv.w);
}

// Per-channel serial scan over T: s = sigmoid(w)*s + k*v.
// Emits bf16 r*s (for Wo GEMM) and bf16 s into hcat[:,1024:].
__global__ __launch_bounds__(256) void scan_k(
    const float* __restrict__ kk, const float* __restrict__ vv,
    const float* __restrict__ rr, const float* __restrict__ wdec,
    unsigned short* __restrict__ rs, unsigned short* __restrict__ hcat)
{
  int id = blockIdx.x * 256 + threadIdx.x;   // 0..2047 = (b, c)
  int b = id >> 10, c = id & 1023;
  float w = sigmf(wdec[c]);
  float s = 0.f;
  for (int t = 0; t < 2048; t++) {
    size_t row = (size_t)(b * 2048 + t);
    size_t idx = row * 1024 + c;
    s = w * s + kk[idx] * vv[idx];
    rs[idx] = f2bf(rr[idx] * s);
    hcat[row * 2048 + 1024 + c] = f2bf(s);
  }
}

__global__ void route_count_k(const int* __restrict__ wnr, int* __restrict__ cnt) {
  int i = blockIdx.x * 256 + threadIdx.x;
  atomicAdd(&cnt[wnr[i * 2 + 0]], 1);
  atomicAdd(&cnt[wnr[i * 2 + 1]], 1);
}
__global__ void route_off_k(const int* __restrict__ cnt, int* __restrict__ off,
                            int* __restrict__ cnt2) {
  int acc = 0;
  for (int e = 0; e < 8; e++) { off[e] = acc; acc += cnt[e]; cnt2[e] = 0; }
}
__global__ void route_assign_k(const int* __restrict__ wnr, const int* __restrict__ off,
                               int* __restrict__ cnt2, int* __restrict__ perm,
                               int* __restrict__ eidx) {
  int i = blockIdx.x * 256 + threadIdx.x;
  for (int j = 0; j < 2; j++) {
    int e   = wnr[i * 2 + j];
    int pos = atomicAdd(&cnt2[e], 1);
    int en  = off[e] + pos;
    perm[en] = i;
    eidx[i * 2 + j] = en;
  }
}
__global__ void combine_k(float* __restrict__ out, const float* __restrict__ slots,
                          const int* __restrict__ eidx) {
  int tok = blockIdx.x, t = threadIdx.x;
  int e0 = eidx[tok * 2 + 0], e1 = eidx[tok * 2 + 1];
#pragma unroll
  for (int i = 0; i < 4; i++) {
    int c = t + i * 256;
    size_t o = (size_t)tok * 1024 + c;
    out[o] += 0.5f * (slots[(size_t)e0 * 1024 + c] + slots[(size_t)e1 * 1024 + c]);
  }
}

extern "C" void kernel_launch(void* const* d_in, const int* in_sizes, int n_in,
                              void* d_out, int out_size, void* d_ws, size_t ws_size,
                              hipStream_t stream)
{
  (void)in_sizes; (void)n_in; (void)out_size; (void)ws_size;
  const float* x       = (const float*)d_in[0];
  const float* v_first = (const float*)d_in[1];
  const int*   winners = (const int*)  d_in[2];
  /* d_in[3] capital_shares: unused (use_market=False) */
  const float* ln1_g = (const float*)d_in[4],  *ln1_b = (const float*)d_in[5];
  const float* ln2_g = (const float*)d_in[6],  *ln2_b = (const float*)d_in[7];
  const float* Wr = (const float*)d_in[8],  *Wk = (const float*)d_in[9];
  const float* Wv = (const float*)d_in[10], *Wo = (const float*)d_in[11];
  const float* wdec = (const float*)d_in[12], *gv = (const float*)d_in[13];
  const float* Wb = (const float*)d_in[14], *bb = (const float*)d_in[15];
  const float* Wk_r = (const float*)d_in[16], *Wv_r = (const float*)d_in[17];
  const float* Wr_r = (const float*)d_in[18];
  const float* W1_t = (const float*)d_in[19], *W2_t = (const float*)d_in[20];
  float* out = (float*)d_out;

  const int BT = 4096, C = 1024, H = 4096, ENT = 8192;

  char* wsp = (char*)d_ws;
  size_t cur = 0;
  auto alloc = [&](size_t bytes) -> void* {
    void* p = wsp + cur; cur += (bytes + 255) & ~(size_t)255; return p;
  };
  unsigned short* h1   = (unsigned short*)alloc((size_t)BT * C * 2);
  unsigned short* rs   = (unsigned short*)alloc((size_t)BT * C * 2);
  unsigned short* hcat = (unsigned short*)alloc((size_t)BT * 2 * C * 2);
  unsigned short* ht   = (unsigned short*)alloc((size_t)BT * C * 2);
  unsigned short* act  = (unsigned short*)alloc((size_t)ENT * H * 2);
  float* rbuf  = (float*)alloc((size_t)BT * C * 4);
  float* kbuf  = (float*)alloc((size_t)BT * C * 4);
  float* vbuf  = (float*)alloc((size_t)BT * C * 4);
  int* cnt  = (int*)alloc(8 * 4);
  int* offd = (int*)alloc(8 * 4);
  int* cnt2 = (int*)alloc(8 * 4);
  int* perm = (int*)alloc(ENT * 4);
  int* eidx = (int*)alloc(ENT * 4);
  // reuse dead buffers: att reuses rbuf (r consumed by scan); slots (32MB)
  // spans kbuf+vbuf (dead after scan, allocated contiguously).
  float* att   = rbuf;
  float* slots = kbuf;

  auto G = [&](int epi, const unsigned short* A, int lda, const float* Bw, int ldb,
               int M, int N, int K, const int* gather, const int* cp, const int* op,
               float* oF, unsigned short* oH, int ldo,
               const float* a0, const float* a1,
               const unsigned short* hs, int ldh, int mtiles) {
    GemmP p{A, lda, Bw, ldb, M, N, K, gather, cp, op, oF, oH, ldo, a0, a1, hs, ldh};
    dim3 grid(N / 128, mtiles), blk(256);
    switch (epi) {
      case EPI_NONE:    gemm_wmma_bf16<EPI_NONE>   <<<grid, blk, 0, stream>>>(p); break;
      case EPI_SIGMOID: gemm_wmma_bf16<EPI_SIGMOID><<<grid, blk, 0, stream>>>(p); break;
      case EPI_VMIX:    gemm_wmma_bf16<EPI_VMIX>   <<<grid, blk, 0, stream>>>(p); break;
      case EPI_BRIDGE:  gemm_wmma_bf16<EPI_BRIDGE> <<<grid, blk, 0, stream>>>(p); break;
      case EPI_RELU2:   gemm_wmma_bf16<EPI_RELU2>  <<<grid, blk, 0, stream>>>(p); break;
      case EPI_GELU:    gemm_wmma_bf16<EPI_GELU>   <<<grid, blk, 0, stream>>>(p); break;
      case EPI_SIGMUL:  gemm_wmma_bf16<EPI_SIGMUL> <<<grid, blk, 0, stream>>>(p); break;
    }
  };

  hipMemsetAsync(cnt, 0, 8 * 4, stream);

  // ---- attention path ----
  ln_k<<<BT, 256, 0, stream>>>(x, nullptr, ln1_g, ln1_b, h1, C, nullptr);
  G(EPI_SIGMOID, h1, C, Wr, C, BT, C, C, 0, 0, 0, rbuf, 0, C, 0, 0, 0, 0, 32);
  G(EPI_NONE,    h1, C, Wk, C, BT, C, C, 0, 0, 0, kbuf, 0, C, 0, 0, 0, 0, 32);
  G(EPI_VMIX,    h1, C, Wv, C, BT, C, C, 0, 0, 0, vbuf, 0, C, v_first, gv, 0, 0, 32);
  scan_k<<<8, 256, 0, stream>>>(kbuf, vbuf, rbuf, wdec, rs, hcat);
  G(EPI_NONE,    rs, C, Wo, C, BT, C, C, 0, 0, 0, att,  0, C, 0, 0, 0, 0, 32);
  ln_k<<<BT, 256, 0, stream>>>(x, att, ln2_g, ln2_b, hcat, 2 * C, out);  // out = x2

  // ---- bridge: ht = bf16(h + [h||state]@Wb + bb) ----
  G(EPI_BRIDGE, hcat, 2 * C, Wb, C, BT, C, 2 * C, 0, 0, 0, 0, ht, C, bb, 0, hcat, 2 * C, 32);

  // ---- top-2 routing (compaction) ----
  route_count_k <<<16, 256, 0, stream>>>(winners, cnt);
  route_off_k   <<<1, 1, 0, stream>>>(cnt, offd, cnt2);
  route_assign_k<<<16, 256, 0, stream>>>(winners, offd, cnt2, perm, eidx);

  // ---- sparse experts ----
  for (int e = 0; e < 6; e++) {   // RWKV FFN: sigmoid(hWr_r) * (relu(hWk_r)^2 Wv_r)
    G(EPI_RELU2,  hcat, 2 * C, Wk_r + (size_t)e * C * H, H, 0, H, C,
      perm, cnt + e, offd + e, 0, act, H, 0, 0, 0, 0, 64);
    G(EPI_NONE,   act, H, Wv_r + (size_t)e * H * C, C, 0, C, H,
      0, cnt + e, offd + e, slots, 0, C, 0, 0, 0, 0, 64);
    G(EPI_SIGMUL, hcat, 2 * C, Wr_r + (size_t)e * C * C, C, 0, C, C,
      perm, cnt + e, offd + e, slots, 0, C, 0, 0, 0, 0, 64);
  }
  for (int te = 0; te < 2; te++) {  // transformer: gelu(ht W1) W2
    G(EPI_GELU, ht, C, W1_t + (size_t)te * C * H, H, 0, H, C,
      perm, cnt + 6 + te, offd + 6 + te, 0, act, H, 0, 0, 0, 0, 64);
    G(EPI_NONE, act, H, W2_t + (size_t)te * H * C, C, 0, C, H,
      0, cnt + 6 + te, offd + 6 + te, slots, 0, C, 0, 0, 0, 0, 64);
  }

  // ---- out = x2 + 0.5*(slot[w0] + slot[w1]) ----
  combine_k<<<BT, 256, 0, stream>>>(out, slots, eidx);
}
